// Attention_76854144795156
// MI455X (gfx1250) — matrix operand order repaired and checked by reading, not compiled, not run on hardware
//
#include <hip/hip_runtime.h>

typedef __attribute__((ext_vector_type(16))) _Float16 v16h;
typedef __attribute__((ext_vector_type(8)))  float    v8f;
typedef int v4i __attribute__((vector_size(4 * sizeof(int))));

constexpr int Bn = 64, Tn = 512, Dn = 1024, Sn = 1024;
constexpr int NU  = 10;        // hidden units
constexpr int NUP = 16;        // padded to WMMA N
constexpr int KB  = Dn / 32;   // 32 k-blocks of 32

// ---------------------------------------------------------------------------
// Repack W1[0:Dn, 0:10] (row-major f32) into f16 WMMA B-fragments.
// B-fragment layout (16x16x32 f16, wave32): lane n<16 -> col u=n, K=0..7,16..23
// packed 2-per-VGPR (low half = even K); lanes 16..31 -> col u=n-16, K=8..15,24..31.
// Stored as w1p[kb*256 + lane*8 + j] (one dword = 2 halves).
// ---------------------------------------------------------------------------
__global__ void pack_w1_kernel(const float* __restrict__ W1,
                               unsigned int* __restrict__ w1p) {
  int idx  = blockIdx.x * blockDim.x + threadIdx.x;   // 0..8191
  int kb   = idx >> 8;
  int rem  = idx & 255;
  int lane = rem >> 3;
  int j    = rem & 7;
  int u    = lane & 15;
  int k    = kb * 32 + ((lane & 16) ? 8 : 0) + 2 * j + ((j & 4) ? 8 : 0);
  float f0 = (u < NU) ? W1[(size_t)k * NU + u]       : 0.0f;
  float f1 = (u < NU) ? W1[(size_t)(k + 1) * NU + u] : 0.0f;
  union { _Float16 h[2]; unsigned int w; } p;
  p.h[0] = (_Float16)f0;
  p.h[1] = (_Float16)f1;
  w1p[idx] = p.w;
}

// ---------------------------------------------------------------------------
// sv[b,u] = b1[u] + sum_s s_prev[b,s] * W1[Dn+s, u]   (padded to 16 with 0)
// ---------------------------------------------------------------------------
__global__ void sv_kernel(const float* __restrict__ s_prev,
                          const float* __restrict__ W1,
                          const float* __restrict__ b1,
                          float* __restrict__ sv) {
  __shared__ float red[NU][256];
  int b = blockIdx.x, tid = threadIdx.x;
  float acc[NU];
#pragma unroll
  for (int u = 0; u < NU; ++u) acc[u] = 0.0f;
  for (int s = tid; s < Sn; s += 256) {
    float sp = s_prev[(size_t)b * Sn + s];
    const float* wrow = W1 + (size_t)(Dn + s) * NU;
#pragma unroll
    for (int u = 0; u < NU; ++u) acc[u] = fmaf(sp, wrow[u], acc[u]);
  }
#pragma unroll
  for (int u = 0; u < NU; ++u) red[u][tid] = acc[u];
  __syncthreads();
  for (int st = 128; st > 0; st >>= 1) {
    if (tid < st) {
#pragma unroll
      for (int u = 0; u < NU; ++u) red[u][tid] += red[u][tid + st];
    }
    __syncthreads();
  }
  if (tid < NUP) sv[b * NUP + tid] = (tid < NU) ? (red[tid][0] + b1[tid]) : 0.0f;
}

// ---------------------------------------------------------------------------
// Scoring kernel: grid (Bn, 4), 256 threads = 8 waves. Wave w handles the
// 16-row t-tile (blockIdx.y*8 + w). Contract D=1024 with WMMA f16 (f32 acc).
// W1 fragments staged into LDS via the CDNA5 async memory->LDS path
// (GLOBAL_LOAD_ASYNC_TO_LDS_B128, ASYNCcnt) when the builtin is available.
// Epilogue: tanh(+sv) dot W2 via 16-lane shfl reduction, relu -> e[b,t].
// ---------------------------------------------------------------------------
__global__ void __launch_bounds__(256) escore_kernel(
    const float* __restrict__ A, const unsigned int* __restrict__ w1p,
    const float* __restrict__ sv, const float* __restrict__ W2,
    const float* __restrict__ b2, float* __restrict__ Eout) {
  __shared__ __align__(32) unsigned int w1s[KB * 256];   // 32 KB B-fragments
  int tid = threadIdx.x;

#if __has_builtin(__builtin_amdgcn_global_load_async_to_lds_b128) && \
    __has_builtin(__builtin_amdgcn_s_wait_asynccnt)
  {
    v4i* g = (v4i*)const_cast<unsigned int*>(w1p);
    v4i* l = (v4i*)w1s;
    // 32 KB total, 16 B per lane per op: 8 async b128 ops per thread.
#pragma unroll
    for (int i = 0; i < 8; ++i) {
      int idx = tid + i * 256;               // units of 16 bytes
      __builtin_amdgcn_global_load_async_to_lds_b128(g + idx, l + idx, 0, 0);
    }
    __builtin_amdgcn_s_wait_asynccnt(0);     // own async ops done before barrier
  }
#else
  for (int i = tid; i < KB * 256; i += 256) w1s[i] = w1p[i];
#endif
  __syncthreads();

  int b    = blockIdx.x;
  int wave = tid >> 5;
  int lane = tid & 31;
  int tile = blockIdx.y * 8 + wave;          // 0..31
  int t0   = tile * 16;
  int trow = t0 + (lane & 15);
  int koff = (lane & 16) ? 8 : 0;            // A-layout half-wave K offset
  const float* arow = A + ((size_t)b * Tn + trow) * Dn + koff;

  v8f acc = {};
#pragma unroll 4
  for (int kb = 0; kb < KB; ++kb) {
    const float4* p = (const float4*)(arow + kb * 32);
    float4 f0 = p[0];        // K = koff + 0..3
    float4 f1 = p[1];        // K = koff + 4..7
    float4 f2 = p[4];        // K = koff + 16..19
    float4 f3 = p[5];        // K = koff + 20..23
    v16h af;
    af[0]  = (_Float16)f0.x; af[1]  = (_Float16)f0.y;
    af[2]  = (_Float16)f0.z; af[3]  = (_Float16)f0.w;
    af[4]  = (_Float16)f1.x; af[5]  = (_Float16)f1.y;
    af[6]  = (_Float16)f1.z; af[7]  = (_Float16)f1.w;
    af[8]  = (_Float16)f2.x; af[9]  = (_Float16)f2.y;
    af[10] = (_Float16)f2.z; af[11] = (_Float16)f2.w;
    af[12] = (_Float16)f3.x; af[13] = (_Float16)f3.y;
    af[14] = (_Float16)f3.z; af[15] = (_Float16)f3.w;
    v16h bf = *(const v16h*)(&w1s[(kb * 32 + lane) * 8]);
    acc = __builtin_amdgcn_wmma_f32_16x16x32_f16(
        /*neg_a=*/false, af, /*neg_b=*/false, bf,
        /*c_mod=*/(short)0, acc, /*reuse_a=*/false, /*reuse_b=*/false);
  }

  int   u    = lane & 15;
  float svu  = sv[b * NUP + u];               // includes b1, 0-padded
  float w2u  = (u < NU) ? W2[u] : 0.0f;
  float b2v  = b2[0];
#pragma unroll
  for (int r = 0; r < 8; ++r) {
    float v = tanhf(acc[r] + svu) * w2u;
    v += __shfl_xor(v, 1, 32);
    v += __shfl_xor(v, 2, 32);
    v += __shfl_xor(v, 4, 32);
    v += __shfl_xor(v, 8, 32);               // sum over 16-lane half (over u)
    if ((lane & 15) == 0) {
      int m = (lane & 16) ? (8 + r) : r;     // C layout: high half = rows 8..15
      Eout[(size_t)b * Tn + t0 + m] = fmaxf(v + b2v, 0.0f);
    }
  }
}

// ---------------------------------------------------------------------------
// In-place softmax over T=512 per batch (scores live in d_out score region).
// ---------------------------------------------------------------------------
__global__ void __launch_bounds__(512) softmax_kernel(float* __restrict__ E) {
  __shared__ float sm[Tn];
  int b = blockIdx.x, tid = threadIdx.x;
  float* Eb = E + (size_t)b * Tn;
  float ev = Eb[tid];
  sm[tid] = ev;
  __syncthreads();
  for (int s = Tn / 2; s > 0; s >>= 1) {
    if (tid < s) sm[tid] = fmaxf(sm[tid], sm[tid + s]);
    __syncthreads();
  }
  float mx = sm[0];
  __syncthreads();
  float ex = __expf(ev - mx);
  sm[tid] = ex;
  __syncthreads();
  for (int s = Tn / 2; s > 0; s >>= 1) {
    if (tid < s) sm[tid] += sm[tid + s];
    __syncthreads();
  }
  Eb[tid] = ex / sm[0];
}

// ---------------------------------------------------------------------------
// context[b,d] = sum_t score[b,t] * a[b,t,d]; grid (Bn, Dn/128) x 128.
// Second read of `a` should hit the 192MB L2.
// ---------------------------------------------------------------------------
__global__ void __launch_bounds__(128) context_kernel(
    const float* __restrict__ A, const float* __restrict__ S,
    float* __restrict__ ctx) {
  int b = blockIdx.x;
  int d = blockIdx.y * 128 + threadIdx.x;
  const float* Sb = S + (size_t)b * Tn;
  const float* Ab = A + (size_t)b * Tn * Dn + d;
  float c = 0.0f;
#pragma unroll 8
  for (int t = 0; t < Tn; ++t) c = fmaf(Sb[t], Ab[(size_t)t * Dn], c);
  ctx[(size_t)b * Dn + d] = c;
}

extern "C" void kernel_launch(void* const* d_in, const int* in_sizes, int n_in,
                              void* d_out, int out_size, void* d_ws, size_t ws_size,
                              hipStream_t stream) {
  const float* a      = (const float*)d_in[0];
  const float* s_prev = (const float*)d_in[1];
  const float* W1     = (const float*)d_in[2];
  const float* b1     = (const float*)d_in[3];
  const float* W2     = (const float*)d_in[4];
  const float* b2     = (const float*)d_in[5];

  float* out    = (float*)d_out;
  float* scores = out + (size_t)Bn * Dn;          // [B,T] region of d_out

  float*        sv  = (float*)d_ws;                                   // 4 KB
  unsigned int* w1p = (unsigned int*)((char*)d_ws +
                                      (size_t)Bn * NUP * sizeof(float)); // 32 KB

  pack_w1_kernel<<<KB, 256, 0, stream>>>(W1, w1p);
  sv_kernel<<<Bn, 256, 0, stream>>>(s_prev, W1, b1, sv);
  escore_kernel<<<dim3(Bn, 4), 256, 0, stream>>>(a, w1p, sv, W2, b2, scores);
  softmax_kernel<<<Bn, Tn, 0, stream>>>(scores);
  context_kernel<<<dim3(Bn, Dn / 128), 128, 0, stream>>>(a, scores, out);
}
